// StackedNNUE_11596411699434
// MI455X (gfx1250) — matrix operand order repaired and checked by reading, not compiled, not run on hardware
//
#include <hip/hip_runtime.h>
#include <hip/hip_bf16.h>
#include <stdint.h>

// ---------------- constants ----------------
#define NSAMP  4096
#define NEXP   32
#define NSLOT  4576            // 4096 + 32*15 worst-case per-expert 16-padding
#define NTILE  286             // NSLOT / 16
#define KFEAT  2240
#define H1DIM  256
#define H2DIM  32
#define NOUTC  51
#define NBLOCK 12

typedef __attribute__((ext_vector_type(16))) _Float16 v16h;
typedef __attribute__((ext_vector_type(8)))  _Float16 v8h;
typedef __attribute__((ext_vector_type(8)))  float    v8f;
typedef unsigned int v4u __attribute__((ext_vector_type(4)));
typedef int          v8i __attribute__((ext_vector_type(8)));
typedef int          v4i __attribute__((ext_vector_type(4)));

// TDM available: ROCm 7.2 (clang-22) 5-arg builtin, gfx1250 device pass only.
#if defined(__gfx1250__) && defined(__has_builtin)
#  if __has_builtin(__builtin_amdgcn_tensor_load_to_lds) && (__clang_major__ < 23)
#    define USE_TDM 1
#  endif
#endif
#ifndef USE_TDM
#define USE_TDM 0
#endif

// ---------------- small device helpers ----------------
__device__ __forceinline__ v8f zero8() {
    v8f z;
#pragma unroll
    for (int i = 0; i < 8; ++i) z[i] = 0.0f;
    return z;
}

__device__ __forceinline__ float gelu_exact(float x) {
    return 0.5f * x * (1.0f + erff(x * 0.70710678118654752f));
}

// butterfly reductions inside each 16-lane half of the wave32
__device__ __forceinline__ float hsum16(float v) {
#pragma unroll
    for (int m = 1; m < 16; m <<= 1) v += __shfl_xor(v, m, 32);
    return v;
}
__device__ __forceinline__ float hmax16(float v) {
#pragma unroll
    for (int m = 1; m < 16; m <<= 1) v = fmaxf(v, __shfl_xor(v, m, 32));
    return v;
}

// WMMA wrapper: D = A(16x32 f16) * B(32x16 f16) + C(16x16 f32)
__device__ __forceinline__ v8f wmma_f16(v16h a, v16h b, v8f c) {
    return __builtin_amdgcn_wmma_f32_16x16x32_f16(
        /*neg_a=*/false, a, /*neg_b=*/false, b,
        /*c_mod=*/(short)0, c, /*reuse_a=*/false, /*reuse_b=*/false);
}

// A fragment (16x32 f16) from row-major f16 matrix, row stride ld (elements).
// ISA layout: lane L: m=L&15, half=L>>4; a[0..7]=K=half*8+i, a[8..15]=K=16+half*8+i
__device__ __forceinline__ v16h load_afrag(const _Float16* base, int ld, int lane) {
    const int m = lane & 15, hl = lane >> 4;
    const _Float16* p = base + (size_t)m * ld + hl * 8;
    v8h lo = *(const v8h*)p;
    v8h hi = *(const v8h*)(p + 16);
    v16h a;
#pragma unroll
    for (int i = 0; i < 8; ++i) { a[i] = lo[i]; a[8 + i] = hi[i]; }
    return a;
}

// B fragment (32x16 f16) from packed n-major layout Bp[n][k], stride ldk.
// ISA layout: lane L: n=L&15, half=L>>4; b[0..7]=K=half*8+i, b[8..15]=K=16+half*8+i
__device__ __forceinline__ v16h load_bfrag(const _Float16* basep, int ldk, int lane) {
    const int n = lane & 15, hl = lane >> 4;
    const _Float16* p = basep + (size_t)n * ldk + hl * 8;
    v8h lo = *(const v8h*)p;
    v8h hi = *(const v8h*)(p + 16);
    v16h b;
#pragma unroll
    for (int i = 0; i < 8; ++i) { b[i] = lo[i]; b[8 + i] = hi[i]; }
    return b;
}

// Build A fragment from a 16x32 f32 LDS tile (optionally through exact GELU)
__device__ __forceinline__ v16h afrag_from_lds(const float (*M)[33], int lane, bool do_gelu) {
    const int m = lane & 15, hl = lane >> 4;
    v16h a;
#pragma unroll
    for (int i = 0; i < 8; ++i) {
        float x0 = M[m][hl * 8 + i];
        float x1 = M[m][16 + hl * 8 + i];
        if (do_gelu) { x0 = gelu_exact(x0); x1 = gelu_exact(x1); }
        a[i]     = (_Float16)x0;
        a[8 + i] = (_Float16)x1;
    }
    return a;
}

#if USE_TDM
// One-shot TDM: copy a contiguous run of f16 (one "line" tile) global -> LDS.
// D# per CDNA5 ISA ch.8: group0 {count=1, lds_addr, global_addr, type=2},
// group1 {data_size=2B, tensor_dim0=nelem, tensor_dim1=1, tile_dim0=nelem,
//         tile_dim1=1, dim0_stride=nelem}; groups 2/3 zero (<=2D tensor).
__device__ __forceinline__ void tdm_load_1d(unsigned lds_off, const void* gsrc,
                                            unsigned nelem_f16) {
    unsigned long long ga = (unsigned long long)(uintptr_t)gsrc;
    v4u g0;
    g0[0] = 1u;                                        // count=1, user mode
    g0[1] = lds_off;                                   // lds_addr (bytes)
    g0[2] = (unsigned)(ga & 0xFFFFFFFFu);              // global_addr[31:0]
    g0[3] = (unsigned)((ga >> 32) & 0x01FFFFFFu)       // global_addr[56:32]
          | 0x80000000u;                               // type=2 ("image")
    v8i g1;
    g1[0] = 0x00010000;                                // data_size=1 (2 bytes)
    g1[1] = (int)((nelem_f16 & 0xFFFFu) << 16);        // tensor_dim0[15:0]
    g1[2] = (int)(((nelem_f16 >> 16) & 0xFFFFu)        // tensor_dim0[31:16]
          | (1u << 16));                               // tensor_dim1 = 1
    g1[3] = (int)((nelem_f16 & 0xFFFFu) << 16);        // tile_dim0
    g1[4] = 1;                                         // tile_dim1 = 1
    g1[5] = (int)nelem_f16;                            // tensor_dim0_stride lo
    g1[6] = 0;
    g1[7] = 0;
    v4i g2 = {0, 0, 0, 0};
    v4i g3 = {0, 0, 0, 0};
    __builtin_amdgcn_tensor_load_to_lds(g0, g1, g2, g3, 0);
}
#endif

// ---------------- sort-by-expert kernels ----------------
__global__ void k_init(int* counts, int* perm) {
    int i = blockIdx.x * blockDim.x + threadIdx.x;
    if (i < NSLOT) perm[i] = -1;
    if (i < NEXP)  counts[i] = 0;
}

__global__ void k_count(const int* __restrict__ desc, int* counts) {
    int i = blockIdx.x * blockDim.x + threadIdx.x;
    if (i < NSAMP) atomicAdd(&counts[desc[i]], 1);
}

__global__ void k_offsets(const int* __restrict__ counts, int* offs, int* cursor) {
    if (blockIdx.x == 0 && threadIdx.x == 0) {
        int acc = 0;
        for (int e = 0; e < NEXP; ++e) {
            offs[e] = acc; cursor[e] = acc;
            acc += (counts[e] + 15) & ~15;      // 16-row WMMA tile alignment
        }
        offs[NEXP] = acc;
    }
}

__global__ void k_scatter(const int* __restrict__ desc, int* cursor, int* perm) {
    int i = blockIdx.x * blockDim.x + threadIdx.x;
    if (i < NSAMP) {
        int e = desc[i];
        int p = atomicAdd(&cursor[e], 1);
        perm[p] = i;
    }
}

__global__ void k_sexp(const int* __restrict__ offs, int* sexp) {
    int s = blockIdx.x * blockDim.x + threadIdx.x;
    if (s < NSLOT) {
        int e = NEXP - 1;
        for (int j = 0; j < NEXP; ++j)
            if (s >= offs[j] && s < offs[j + 1]) e = j;
        sexp[s] = e;
    }
}

// ---------------- feature build (sorted f16 rows) ----------------
__global__ __launch_bounds__(64) void k_build_x(
    const int* __restrict__ piece_idx, const int* __restrict__ side_flag,
    const int* __restrict__ ep_file,   const float* __restrict__ castle_ms,
    const float* __restrict__ fifty_a,
    const float* __restrict__ Wp_w, const float* __restrict__ Wp_b,
    const float* __restrict__ Wc_w, const float* __restrict__ Wc_b,
    const float* __restrict__ Wep_w, const float* __restrict__ Wep_b,
    const float* __restrict__ Wf_w, const float* __restrict__ Wf_b,
    const int* __restrict__ perm, const int* __restrict__ sexp,
    _Float16* __restrict__ Xs)
{
    const int slot = blockIdx.x;
    const int tid  = threadIdx.x;
    _Float16* xrow = Xs + (size_t)slot * KFEAT;
    const int smp = perm[slot];
    if (smp < 0) {                       // padding slot: zero row
        for (int j = tid; j < KFEAT; j += 64) xrow[j] = (_Float16)0.0f;
        return;
    }
    const int e    = sexp[slot];         // == descriptor_index[smp]
    const int side = side_flag[smp];

    // pieces: one square per thread (64 threads, 64 squares), 32 feats each
    {
        const int sq = tid;
        const int pc = piece_idx[smp * 64 + sq];
        const float msk = (pc >= 0) ? 1.0f : 0.0f;
        const int pcc = pc < 0 ? 0 : pc;
        const float* tab = (side ? Wp_b : Wp_w) + (((size_t)e * 64 + sq) * 12 + pcc) * 32;
#pragma unroll 4
        for (int k = 0; k < 32; ++k)
            xrow[sq * 32 + k] = (_Float16)(tab[k] * msk);
    }
    // castle (128) + ep (32) + fifty (32)
    for (int j = tid; j < 192; j += 64) {
        float v;
        if (j < 128) {
            const int a = j >> 5, k = j & 31;
            const float* Wc = (side ? Wc_b : Wc_w) + ((size_t)e * 4 + a) * 32;
            v = castle_ms[smp * 4 + a] * Wc[k];
        } else if (j < 160) {
            const int k = j - 128;
            const int ep = ep_file[smp];
            const float msk = (ep >= 0) ? 1.0f : 0.0f;
            const int epc = ep < 0 ? 0 : ep;
            v = (side ? Wep_b : Wep_w)[((size_t)e * 8 + epc) * 32 + k] * msk;
        } else {
            const int k = j - 160;
            const float aa = fifty_a[smp];
            const float* Wf = (side ? Wf_b : Wf_w) + (size_t)e * 64;
            v = (1.0f - aa) * Wf[k] + aa * Wf[32 + k];
        }
        xrow[2048 + j] = (_Float16)v;
    }
}

// ---------------- weight repack f32 -> f16 (B-operand n-major layout) -------
__global__ __launch_bounds__(256) void k_conv_fc1(const float* __restrict__ W,
                                                  _Float16* __restrict__ Wp) {
    const int n  = threadIdx.x;        // 0..255  (coalesced source reads over n)
    const int e  = blockIdx.y;         // 0..31
    const int k0 = blockIdx.x * 8;     // 0..2232
    const float* src = W + ((size_t)e * KFEAT) * H1DIM + n;
    _Float16* dst = Wp + ((size_t)e * H1DIM + n) * KFEAT + k0;
#pragma unroll
    for (int k = 0; k < 8; ++k)
        dst[k] = (_Float16)src[(size_t)(k0 + k) * H1DIM];
}

__global__ __launch_bounds__(256) void k_conv_small(
    const float* __restrict__ fc2_w, const float* __restrict__ out_w,
    const float* __restrict__ blkA_w, const float* __restrict__ blkB_w,
    _Float16* __restrict__ fc2p, _Float16* __restrict__ outp,
    _Float16* __restrict__ blkAp, _Float16* __restrict__ blkBp)
{
    const int SZ_FC2 = NEXP * H2DIM * H1DIM;   // 262144  [e][n32][k256]
    const int SZ_OUT = NEXP * 64 * H2DIM;      // 65536   [e][n64][k32] (51 padded)
    const int SZ_BLK = NBLOCK * NEXP * 32 * 32;// 393216  [b*e][n32][k32]
    int id = blockIdx.x * 256 + threadIdx.x;
    if (id < SZ_FC2) {
        int e = id / (H2DIM * H1DIM), rem = id % (H2DIM * H1DIM);
        int nn = rem / H1DIM, k = rem % H1DIM;
        fc2p[id] = (_Float16)fc2_w[((size_t)e * H1DIM + k) * H2DIM + nn];
        return;
    }
    id -= SZ_FC2;
    if (id < SZ_OUT) {
        int e = id / (64 * 32), rem = id % (64 * 32);
        int nn = rem / 32, k = rem % 32;
        outp[(size_t)e * 64 * 32 + rem] =
            (nn < NOUTC) ? (_Float16)out_w[((size_t)e * 32 + k) * NOUTC + nn]
                         : (_Float16)0.0f;
        return;
    }
    id -= SZ_OUT;
    if (id < SZ_BLK) {
        int be = id / 1024, rem = id % 1024;
        int nn = rem / 32, k = rem % 32;
        blkAp[id] = (_Float16)blkA_w[((size_t)be * 32 + k) * 32 + nn];
        return;
    }
    id -= SZ_BLK;
    if (id < SZ_BLK) {
        int be = id / 1024, rem = id % 1024;
        int nn = rem / 32, k = rem % 32;
        blkBp[id] = (_Float16)blkB_w[((size_t)be * 32 + k) * 32 + nn];
    }
}

// ---------------- fc1: 2240 -> 256 GEMM + GELU + LayerNorm -------------------
// grid.x = 286 row tiles, block = 128 (4 waves, each wave owns 64 output cols).
// A tile (16x2240 f16, 71680B) staged in LDS once per block via TDM; B fragments
// ping-pong double-buffered (2x manual unroll, no register rotation copies).
__global__ __launch_bounds__(128) void k_fc1(
    const _Float16* __restrict__ Xs, const _Float16* __restrict__ W1p,
    const float* __restrict__ fc1_b, const float* __restrict__ ln1_g,
    const float* __restrict__ ln1_b, const int* __restrict__ sexp,
    _Float16* __restrict__ h1)
{
    __shared__ __align__(16) _Float16 sA[16 * KFEAT];
    __shared__ float sm_s[4][16], sm_q[4][16];

    const int tile = blockIdx.x;
    const int lane = threadIdx.x & 31;
    const int wave = threadIdx.x >> 5;
    const int e    = sexp[tile * 16];
    const int col0 = wave * 64;
    const int n  = lane & 15, hl = lane >> 4;

    const _Float16* Asrc = Xs + (size_t)tile * 16 * KFEAT;
#if USE_TDM
    if (wave == 0) {
        tdm_load_1d((unsigned)(uintptr_t)sA, Asrc, 16 * KFEAT);
        __builtin_amdgcn_s_wait_tensorcnt(0);
    }
#else
    {
        const v8h* src = (const v8h*)Asrc;
        v8h* dst = (v8h*)sA;
        for (int i = threadIdx.x; i < 16 * KFEAT / 8; i += 128) dst[i] = src[i];
    }
#endif
    __syncthreads();

    const _Float16* Bbase = W1p + (size_t)e * H1DIM * KFEAT;
    v8f acc[4];
#pragma unroll
    for (int t = 0; t < 4; ++t) acc[t] = zero8();

    // ping-pong double-buffered B fragments: b0 = even K-steps, b1 = odd.
    v16h b0[4], b1[4];
#pragma unroll
    for (int t = 0; t < 4; ++t)
        b0[t] = load_bfrag(Bbase + (size_t)(col0 + t * 16) * KFEAT, KFEAT, lane);

    int kk = 0;
    for (; kk + 64 < KFEAT; kk += 64) {     // 34 iterations (2 K-steps each)
        __builtin_prefetch(Bbase + (size_t)col0 * KFEAT + kk + 96, 0, 1);
#pragma unroll
        for (int t = 0; t < 4; ++t)
            b1[t] = load_bfrag(Bbase + (size_t)(col0 + t * 16) * KFEAT + kk + 32,
                               KFEAT, lane);
        {
            v16h a = load_afrag(sA + kk, KFEAT, lane);
#pragma unroll
            for (int t = 0; t < 4; ++t) acc[t] = wmma_f16(a, b0[t], acc[t]);
        }
#pragma unroll
        for (int t = 0; t < 4; ++t)
            b0[t] = load_bfrag(Bbase + (size_t)(col0 + t * 16) * KFEAT + kk + 64,
                               KFEAT, lane);
        {
            v16h a = load_afrag(sA + kk + 32, KFEAT, lane);
#pragma unroll
            for (int t = 0; t < 4; ++t) acc[t] = wmma_f16(a, b1[t], acc[t]);
        }
    }
    {   // tail: kk == KFEAT-64; b0 already holds fragments for this K-step
#pragma unroll
        for (int t = 0; t < 4; ++t)
            b1[t] = load_bfrag(Bbase + (size_t)(col0 + t * 16) * KFEAT + kk + 32,
                               KFEAT, lane);
        v16h a = load_afrag(sA + kk, KFEAT, lane);
#pragma unroll
        for (int t = 0; t < 4; ++t) acc[t] = wmma_f16(a, b0[t], acc[t]);
        a = load_afrag(sA + kk + 32, KFEAT, lane);
#pragma unroll
        for (int t = 0; t < 4; ++t) acc[t] = wmma_f16(a, b1[t], acc[t]);
    }

    // bias + GELU, accumulate per-row stats
    float s[8], q[8];
#pragma unroll
    for (int r = 0; r < 8; ++r) { s[r] = 0.0f; q[r] = 0.0f; }
#pragma unroll
    for (int t = 0; t < 4; ++t) {
        const int col = col0 + t * 16 + n;
        const float bias = fc1_b[e * H1DIM + col];
#pragma unroll
        for (int r = 0; r < 8; ++r) {
            float x = acc[t][r] + bias;
            x = gelu_exact(x);
            acc[t][r] = x;
            s[r] += x; q[r] += x * x;
        }
    }
#pragma unroll
    for (int r = 0; r < 8; ++r) { s[r] = hsum16(s[r]); q[r] = hsum16(q[r]); }

    if (n == 0) {
#pragma unroll
        for (int r = 0; r < 8; ++r) {
            sm_s[wave][hl * 8 + r] = s[r];
            sm_q[wave][hl * 8 + r] = q[r];
        }
    }
    __syncthreads();

    float mean[8], rstd[8];
#pragma unroll
    for (int r = 0; r < 8; ++r) {
        const int row = r + 8 * hl;
        float ts = 0.0f, tq = 0.0f;
#pragma unroll
        for (int w = 0; w < 4; ++w) { ts += sm_s[w][row]; tq += sm_q[w][row]; }
        const float m = ts * (1.0f / 256.0f);
        const float v = tq * (1.0f / 256.0f) - m * m;
        mean[r] = m;
        rstd[r] = rsqrtf(v + 1e-5f);
    }

#pragma unroll
    for (int t = 0; t < 4; ++t) {
        const int col = col0 + t * 16 + n;
        const float g  = ln1_g[e * H1DIM + col];
        const float bb = ln1_b[e * H1DIM + col];
#pragma unroll
        for (int r = 0; r < 8; ++r) {
            const int row = r + 8 * hl;
            const float y = (acc[t][r] - mean[r]) * rstd[r] * g + bb;
            h1[((size_t)tile * 16 + row) * H1DIM + col] = (_Float16)y;
        }
    }
}

// ---------------- fused tail: fc2 + 12 residual blocks + head + softmax ------
// post-GEMM helper for a 16x32 result in two D tiles: bias+GELU+LayerNorm(32)
__device__ __forceinline__ void post32(v8f* acc, const float* bias, const float* g,
                                       const float* bb, float (*Dst)[33],
                                       int lane, bool add_to_dst) {
    const int n = lane & 15, hl = lane >> 4;
    float s[8], q[8];
#pragma unroll
    for (int r = 0; r < 8; ++r) { s[r] = 0.0f; q[r] = 0.0f; }
#pragma unroll
    for (int t = 0; t < 2; ++t) {
        const float bi = bias[t * 16 + n];
#pragma unroll
        for (int r = 0; r < 8; ++r) {
            float x = acc[t][r] + bi;
            x = gelu_exact(x);
            acc[t][r] = x;
            s[r] += x; q[r] += x * x;
        }
    }
#pragma unroll
    for (int r = 0; r < 8; ++r) { s[r] = hsum16(s[r]); q[r] = hsum16(q[r]); }
#pragma unroll
    for (int t = 0; t < 2; ++t) {
        const float gg  = g[t * 16 + n];
        const float bbv = bb[t * 16 + n];
#pragma unroll
        for (int r = 0; r < 8; ++r) {
            const int row = r + 8 * hl;
            const float m  = s[r] * (1.0f / 32.0f);
            const float vv = q[r] * (1.0f / 32.0f) - m * m;
            const float y = (acc[t][r] - m) * rsqrtf(vv + 1e-5f) * gg + bbv;
            if (add_to_dst) Dst[row][t * 16 + n] += y;
            else            Dst[row][t * 16 + n]  = y;
        }
    }
}

__global__ __launch_bounds__(32) void k_tail(
    const _Float16* __restrict__ h1,
    const _Float16* __restrict__ fc2p, const float* __restrict__ fc2_b,
    const float* __restrict__ ln2_g, const float* __restrict__ ln2_b,
    const _Float16* __restrict__ blkAp, const float* __restrict__ blkA_b,
    const float* __restrict__ blkA_g, const float* __restrict__ blkA_be,
    const _Float16* __restrict__ blkBp, const float* __restrict__ blkB_b,
    const float* __restrict__ blkB_g, const float* __restrict__ blkB_be,
    const _Float16* __restrict__ outp, const float* __restrict__ out_b,
    const float* __restrict__ bins,
    const int* __restrict__ sexp, const int* __restrict__ perm,
    float* __restrict__ out_logits, float* __restrict__ out_pwin)
{
    const int tile = blockIdx.x;
    const int lane = threadIdx.x & 31;
    const int n = lane & 15, hl = lane >> 4;
    const int e = sexp[tile * 16];

    __shared__ __align__(16) _Float16 sH1[16 * H1DIM];  // fc2 A tile (8KB)
    __shared__ float H[16][33];   // running hidden (carry), row-major
    __shared__ float Y[16][33];   // intermediate within a block

    const _Float16* Asrc = h1 + (size_t)tile * 16 * H1DIM;
#if USE_TDM
    tdm_load_1d((unsigned)(uintptr_t)sH1, Asrc, 16 * H1DIM);
    __builtin_amdgcn_s_wait_tensorcnt(0);
#else
    {
        const v8h* src = (const v8h*)Asrc;
        v8h* dst = (v8h*)sH1;
        for (int i = threadIdx.x; i < 16 * H1DIM / 8; i += 32) dst[i] = src[i];
    }
#endif
    __syncthreads();

    // ---- fc2: 256 -> 32 (ping-pong double-buffered B fragments) ----
    {
        v8f acc[2] = { zero8(), zero8() };
        const _Float16* B2 = fc2p + (size_t)e * H2DIM * H1DIM;
        v16h c0[2], c1[2];
        c0[0] = load_bfrag(B2,              H1DIM, lane);
        c0[1] = load_bfrag(B2 + 16 * H1DIM, H1DIM, lane);
        int kk = 0;
        for (; kk + 64 < H1DIM; kk += 64) {     // 3 iterations (2 K-steps each)
            c1[0] = load_bfrag(B2 + kk + 32,              H1DIM, lane);
            c1[1] = load_bfrag(B2 + 16 * H1DIM + kk + 32, H1DIM, lane);
            v16h a = load_afrag(sH1 + kk, H1DIM, lane);
            acc[0] = wmma_f16(a, c0[0], acc[0]);
            acc[1] = wmma_f16(a, c0[1], acc[1]);
            c0[0] = load_bfrag(B2 + kk + 64,              H1DIM, lane);
            c0[1] = load_bfrag(B2 + 16 * H1DIM + kk + 64, H1DIM, lane);
            a = load_afrag(sH1 + kk + 32, H1DIM, lane);
            acc[0] = wmma_f16(a, c1[0], acc[0]);
            acc[1] = wmma_f16(a, c1[1], acc[1]);
        }
        {   // tail: kk == H1DIM-64, c0 already loaded for this K-step
            c1[0] = load_bfrag(B2 + kk + 32,              H1DIM, lane);
            c1[1] = load_bfrag(B2 + 16 * H1DIM + kk + 32, H1DIM, lane);
            v16h a = load_afrag(sH1 + kk, H1DIM, lane);
            acc[0] = wmma_f16(a, c0[0], acc[0]);
            acc[1] = wmma_f16(a, c0[1], acc[1]);
            a = load_afrag(sH1 + kk + 32, H1DIM, lane);
            acc[0] = wmma_f16(a, c1[0], acc[0]);
            acc[1] = wmma_f16(a, c1[1], acc[1]);
        }
        post32(acc, fc2_b + e * 32, ln2_g + e * 32, ln2_b + e * 32, H, lane, false);
    }
    __syncthreads();

    // ---- 12 residual blocks ----
    for (int blk = 0; blk < NBLOCK; ++blk) {
        const int be = blk * NEXP + e;
        const _Float16* BA = blkAp + (size_t)be * 32 * 32;
        const _Float16* BB = blkBp + (size_t)be * 32 * 32;

        // layer A: y1 = LN(gelu(H @ wA + bA))
        {
            v16h a = afrag_from_lds(H, lane, false);
            v8f acc[2] = { zero8(), zero8() };
            acc[0] = wmma_f16(a, load_bfrag(BA,           32, lane), acc[0]);
            acc[1] = wmma_f16(a, load_bfrag(BA + 16 * 32, 32, lane), acc[1]);
            post32(acc, blkA_b + be * 32, blkA_g + be * 32, blkA_be + be * 32,
                   Y, lane, false);
        }
        __syncthreads();
        // layer B: H += LN(gelu(y1 @ wB + bB))
        {
            v16h a = afrag_from_lds(Y, lane, false);
            v8f acc[2] = { zero8(), zero8() };
            acc[0] = wmma_f16(a, load_bfrag(BB,           32, lane), acc[0]);
            acc[1] = wmma_f16(a, load_bfrag(BB + 16 * 32, 32, lane), acc[1]);
            post32(acc, blkB_b + be * 32, blkB_g + be * 32, blkB_be + be * 32,
                   H, lane, true);
        }
        __syncthreads();
    }

    // ---- head: logits = gelu(H) @ out_w + out_b  (51 cols padded to 64) ----
    v16h ag = afrag_from_lds(H, lane, true);
    v8f oc[4];
#pragma unroll
    for (int t = 0; t < 4; ++t) {
        oc[t] = zero8();
        oc[t] = wmma_f16(ag, load_bfrag(outp + (size_t)e * 64 * 32 + (size_t)(t * 16) * 32,
                                        32, lane), oc[t]);
    }

    float logit[4][8];
    float mx[8];
#pragma unroll
    for (int r = 0; r < 8; ++r) mx[r] = -1e30f;
#pragma unroll
    for (int t = 0; t < 4; ++t) {
        const int col = t * 16 + n;
        const bool valid = col < NOUTC;
        const float bi = valid ? out_b[e * NOUTC + col] : 0.0f;
#pragma unroll
        for (int r = 0; r < 8; ++r) {
            logit[t][r] = oc[t][r] + bi;
            if (valid) mx[r] = fmaxf(mx[r], logit[t][r]);
        }
    }
#pragma unroll
    for (int r = 0; r < 8; ++r) mx[r] = hmax16(mx[r]);

    float ssum[8], pw[8];
#pragma unroll
    for (int r = 0; r < 8; ++r) { ssum[r] = 0.0f; pw[r] = 0.0f; }
#pragma unroll
    for (int t = 0; t < 4; ++t) {
        const int col = t * 16 + n;
        if (col < NOUTC) {
            const float bv = bins[col];
#pragma unroll
            for (int r = 0; r < 8; ++r) {
                const float ex = expf(logit[t][r] - mx[r]);
                ssum[r] += ex;
                pw[r]   += ex * bv;
            }
        }
    }
#pragma unroll
    for (int r = 0; r < 8; ++r) { ssum[r] = hsum16(ssum[r]); pw[r] = hsum16(pw[r]); }

    // ---- scatter back by original sample index ----
#pragma unroll
    for (int t = 0; t < 4; ++t) {
        const int col = t * 16 + n;
        if (col >= NOUTC) continue;
#pragma unroll
        for (int r = 0; r < 8; ++r) {
            const int row = r + 8 * hl;
            const int smp = perm[tile * 16 + row];
            if (smp >= 0) out_logits[(size_t)smp * NOUTC + col] = logit[t][r];
        }
    }
    if (n == 0) {
#pragma unroll
        for (int r = 0; r < 8; ++r) {
            const int row = r + 8 * hl;
            const int smp = perm[tile * 16 + row];
            if (smp >= 0) out_pwin[smp] = pw[r] / ssum[r];
        }
    }
}

// ---------------- host launcher ----------------
extern "C" void kernel_launch(void* const* d_in, const int* in_sizes, int n_in,
                              void* d_out, int out_size, void* d_ws, size_t ws_size,
                              hipStream_t stream) {
    (void)in_sizes; (void)n_in; (void)out_size; (void)ws_size;

    const int*   piece_idx = (const int*)d_in[0];
    const int*   side_flag = (const int*)d_in[1];
    const int*   ep_file   = (const int*)d_in[2];
    const float* castle_ms = (const float*)d_in[3];
    const float* fifty_a   = (const float*)d_in[4];
    const int*   desc      = (const int*)d_in[5];
    const float* Wp_w = (const float*)d_in[6],  *Wp_b = (const float*)d_in[7];
    const float* Wc_w = (const float*)d_in[8],  *Wc_b = (const float*)d_in[9];
    const float* Wep_w = (const float*)d_in[10], *Wep_b = (const float*)d_in[11];
    const float* Wf_w = (const float*)d_in[12], *Wf_b = (const float*)d_in[13];
    const float* fc1_w = (const float*)d_in[14], *fc1_b = (const float*)d_in[15];
    const float* ln1_g = (const float*)d_in[16], *ln1_b = (const float*)d_in[17];
    const float* fc2_w = (const float*)d_in[18], *fc2_b = (const float*)d_in[19];
    const float* ln2_g = (const float*)d_in[20], *ln2_b = (const float*)d_in[21];
    const float* blkA_w = (const float*)d_in[22], *blkA_b = (const float*)d_in[23];
    const float* blkA_g = (const float*)d_in[24], *blkA_be = (const float*)d_in[25];
    const float* blkB_w = (const float*)d_in[26], *blkB_b = (const float*)d_in[27];
    const float* blkB_g = (const float*)d_in[28], *blkB_be = (const float*)d_in[29];
    const float* out_w = (const float*)d_in[30], *out_b = (const float*)d_in[31];
    const float* bins  = (const float*)d_in[32];

    // carve workspace (256B aligned regions)
    uint8_t* base = (uint8_t*)d_ws;
    size_t cur = 0;
    auto take = [&](size_t bytes) -> void* {
        void* p = base + cur;
        cur = (cur + bytes + 255) & ~(size_t)255;
        return p;
    };
    int* counts = (int*)take(NEXP * sizeof(int));
    int* offs   = (int*)take((NEXP + 1) * sizeof(int));
    int* cursor = (int*)take(NEXP * sizeof(int));
    int* perm   = (int*)take(NSLOT * sizeof(int));
    int* sexp   = (int*)take(NSLOT * sizeof(int));
    _Float16* Xs   = (_Float16*)take((size_t)NSLOT * KFEAT * sizeof(_Float16));
    _Float16* h1   = (_Float16*)take((size_t)NSLOT * H1DIM * sizeof(_Float16));
    _Float16* W1p  = (_Float16*)take((size_t)NEXP * H1DIM * KFEAT * sizeof(_Float16));
    _Float16* fc2p = (_Float16*)take((size_t)NEXP * H2DIM * H1DIM * sizeof(_Float16));
    _Float16* outp = (_Float16*)take((size_t)NEXP * 64 * 32 * sizeof(_Float16));
    _Float16* blkAp = (_Float16*)take((size_t)NBLOCK * NEXP * 32 * 32 * sizeof(_Float16));
    _Float16* blkBp = (_Float16*)take((size_t)NBLOCK * NEXP * 32 * 32 * sizeof(_Float16));

    float* out_logits = (float*)d_out;
    float* out_pwin   = (float*)d_out + (size_t)NSAMP * NOUTC;

    // 1) sort samples by expert with 16-row padding
    k_init   <<<(NSLOT + 255) / 256, 256, 0, stream>>>(counts, perm);
    k_count  <<<(NSAMP + 255) / 256, 256, 0, stream>>>(desc, counts);
    k_offsets<<<1, 32, 0, stream>>>(counts, offs, cursor);
    k_scatter<<<(NSAMP + 255) / 256, 256, 0, stream>>>(desc, cursor, perm);
    k_sexp   <<<(NSLOT + 255) / 256, 256, 0, stream>>>(offs, sexp);

    // 2) weight repack f32 -> packed f16 (independent of the sort)
    k_conv_fc1  <<<dim3(KFEAT / 8, NEXP), 256, 0, stream>>>(fc1_w, W1p);
    {
        const int total = NEXP * H2DIM * H1DIM + NEXP * 64 * 32 + 2 * NBLOCK * NEXP * 32 * 32;
        k_conv_small<<<(total + 255) / 256, 256, 0, stream>>>(
            fc2_w, out_w, blkA_w, blkB_w, fc2p, outp, blkAp, blkBp);
    }

    // 3) build sorted f16 feature rows
    k_build_x<<<NSLOT, 64, 0, stream>>>(
        piece_idx, side_flag, ep_file, castle_ms, fifty_a,
        Wp_w, Wp_b, Wc_w, Wc_b, Wep_w, Wep_b, Wf_w, Wf_b,
        perm, sexp, Xs);

    // 4) fc1 GEMM (WMMA, TDM-staged A, ping-pong B) + GELU + LN
    k_fc1<<<NTILE, 128, 0, stream>>>(Xs, W1p, fc1_b, ln1_g, ln1_b, sexp, h1);

    // 5) fused fc2 + residual blocks + head + softmax, scatter to d_out
    k_tail<<<NTILE, 32, 0, stream>>>(
        h1, fc2p, fc2_b, ln2_g, ln2_b,
        blkAp, blkA_b, blkA_g, blkA_be,
        blkBp, blkB_b, blkB_g, blkB_be,
        outp, out_b, bins, sexp, perm, out_logits, out_pwin);
}
